// unit_gcn_71957882077397
// MI455X (gfx1250) — matrix operand order repaired and checked by reading, not compiled, not run on hardware
//
#include <hip/hip_runtime.h>
#include <math.h>

// ---------------- types for WMMA fragments ----------------
typedef __attribute__((ext_vector_type(16))) __bf16       v16bf;
typedef __attribute__((ext_vector_type(8)))  float        v8f;
typedef __attribute__((ext_vector_type(4)))  unsigned int u32x4;

union ABfrag { u32x4 q[2]; v16bf v; };

__device__ __forceinline__ unsigned short bf16rne(float f) {
    unsigned int u = __float_as_uint(f);
    unsigned int r = u + 0x7FFFu + ((u >> 16) & 1u);
    return (unsigned short)(r >> 16);
}

// Problem constants
#define NN 128
#define CIN 64
#define COUT 64
#define TT 128
#define VV 25
#define SS 3
#define RR 8

// ---------------- 1) xbar[n][c][v] = mean_t x[n][c][t][v] ----------------
__global__ __launch_bounds__(256) void kmean(const float* __restrict__ x,
                                             float* __restrict__ xbar) {
    int id = blockIdx.x * 256 + threadIdx.x;
    if (id >= NN * CIN * VV) return;
    int v = id % VV;
    int q = id / VV;
    int c = q % CIN;
    int n = q / CIN;
    const float* p = x + (((size_t)n * CIN + c) * TT) * VV + v;
    float s = 0.f;
    #pragma unroll 8
    for (int t = 0; t < TT; ++t) s += p[(size_t)t * VV];
    xbar[id] = s * (1.f / (float)TT);
}

// ---------------- 2) adjacency: adjB[s][n][o][u:32][v:32] (bf16, zero-padded)
__global__ __launch_bounds__(256) void kadj(const float* __restrict__ xbar,
                                            const float* __restrict__ W1, const float* __restrict__ b1,
                                            const float* __restrict__ W2, const float* __restrict__ b2,
                                            const float* __restrict__ W4, const float* __restrict__ b4,
                                            const float* __restrict__ PA, const float* __restrict__ alpha,
                                            unsigned short* __restrict__ adjB) {
    __shared__ float xb[CIN * VV];      // 1600
    __shared__ float sx1[RR * VV];      // 200
    __shared__ float sx2[RR * VV];
    __shared__ float sW1[RR * CIN], sW2[RR * CIN];   // 512 each
    __shared__ float sW4[COUT * RR];                 // 512
    __shared__ float sb4[COUT];

    int bid = blockIdx.x;
    int s = bid % SS;
    int n = bid / SS;
    int tid = threadIdx.x;

    for (int e = tid; e < CIN * VV; e += 256) xb[e] = xbar[(size_t)n * (CIN * VV) + e];
    for (int e = tid; e < RR * CIN; e += 256) {
        sW1[e] = W1[s * RR * CIN + e];
        sW2[e] = W2[s * RR * CIN + e];
    }
    for (int e = tid; e < COUT * RR; e += 256) sW4[e] = W4[s * COUT * RR + e];
    if (tid < COUT) sb4[tid] = b4[s * COUT + tid];
    __syncthreads();

    if (tid < RR * VV) {
        int r = tid / VV, v = tid % VV;
        float a1 = b1[s * RR + r], a2 = b2[s * RR + r];
        #pragma unroll 8
        for (int c = 0; c < CIN; ++c) {
            float xv = xb[c * VV + v];
            a1 = fmaf(sW1[r * CIN + c], xv, a1);
            a2 = fmaf(sW2[r * CIN + c], xv, a2);
        }
        sx1[tid] = a1;
        sx2[tid] = a2;
    }
    __syncthreads();

    float al = alpha[s];
    size_t base = (((size_t)s * NN + n) * COUT) * 1024;
    for (int p = tid; p < 1024; p += 256) {   // p = u*32 + v over padded 32x32
        int u = p >> 5, v = p & 31;
        if (u < VV && v < VV) {
            float d[RR];
            #pragma unroll
            for (int r = 0; r < RR; ++r) d[r] = tanhf(sx1[r * VV + u] - sx2[r * VV + v]);
            float pa = PA[s * VV * VV + u * VV + v];
            #pragma unroll 4
            for (int o = 0; o < COUT; ++o) {
                float a = sb4[o];
                #pragma unroll
                for (int r = 0; r < RR; ++r) a = fmaf(sW4[o * RR + r], d[r], a);
                adjB[base + (size_t)o * 1024 + p] = bf16rne(a * al + pa);
            }
        } else {
            for (int o = 0; o < COUT; ++o) adjB[base + (size_t)o * 1024 + p] = 0;
        }
    }
}

// ---------------- 3) main WMMA kernel ----------------
// block = (n, tb): 16 t-rows. LDS: W3 bf16 [3][64][64] | x bf16 [col=400][c=64]
//                               | x3 bf16 [3][64][16][32]
#define SM_W3  0
#define SM_XS  24576
#define SM_X3  75776
#define SM_TOT 272384

__global__ __launch_bounds__(256) void kmain(const float* __restrict__ x,
                                             const float* __restrict__ W3,
                                             const float* __restrict__ b3,
                                             const unsigned short* __restrict__ adjB,
                                             float* __restrict__ y) {
    extern __shared__ char smem[];
    unsigned short* w3s = (unsigned short*)(smem + SM_W3);   // [s][o][c]
    unsigned short* xsb = (unsigned short*)(smem + SM_XS);   // [col][c]
    unsigned short* x3s = (unsigned short*)(smem + SM_X3);   // [s][o][t][v32]

    const int tid  = threadIdx.x;
    const int n    = blockIdx.x >> 3;
    const int tb   = blockIdx.x & 7;
    const int wave = tid >> 5;
    const int lane = tid & 31;
    const int lmod = lane & 15;
    const int lhi  = lane >> 4;       // 0/1 half-wave

    // stage W3 (all subsets) as bf16
    for (int e = tid; e < SS * COUT * CIN; e += 256) w3s[e] = bf16rne(W3[e]);
    // stage x slice [c][t:16][v] -> xsb[col=t*25+v][c]
    for (int e = tid; e < 400 * CIN; e += 256) {
        int c = e / 400, col = e % 400;
        int tl = col / VV, v = col % VV;
        float xv = x[(((size_t)n * CIN + c) * TT + (tb * 16 + tl)) * VV + v];
        xsb[col * CIN + c] = bf16rne(xv);
    }
    // zero the v-padding of x3s (v = 25..31)
    for (int e = tid; e < SS * COUT * 16 * 7; e += 256) {
        int v = VV + e % 7;
        int q = e / 7;
        int t = q % 16; q /= 16;
        int o = q % COUT;
        int s = q / COUT;
        x3s[((s * COUT + o) * 16 + t) * 32 + v] = 0;
    }
    __syncthreads();

    // ---- stage 1: x3[s][o][t][v] = W3[s]*x + b3  (M=o, K=c, N=t*25+v) ----
    for (int s = 0; s < SS; ++s) {
        for (int tile = wave; tile < 100; tile += 8) {
            int otile = tile / 25, coltile = tile % 25;
            v8f acc = {};
            #pragma unroll
            for (int kk = 0; kk < 2; ++kk) {
                ABfrag a, b;
                int o  = otile * 16 + lmod;
                int c1 = kk * 32 + lhi * 16;                // A: linear K per half-wave
                const unsigned short* arow = w3s + (s * COUT + o) * CIN;
                a.q[0] = *(const u32x4*)(arow + c1);
                a.q[1] = *(const u32x4*)(arow + c1 + 8);
                int col = coltile * 16 + lmod;
                int cb  = kk * 32 + lhi * 16;               // B: linear K per half-wave
                const unsigned short* brow = xsb + col * CIN + cb;
                b.q[0] = *(const u32x4*)(brow);
                b.q[1] = *(const u32x4*)(brow + 8);
                acc = __builtin_amdgcn_wmma_f32_16x16x32_bf16(
                        false, a.v, false, b.v, (short)0, acc, false, false);
            }
            int col = coltile * 16 + lmod;
            int tl = col / VV, v = col % VV;
            #pragma unroll
            for (int g = 0; g < 8; ++g) {
                int o = otile * 16 + g + lhi * 8;           // D: M = g + 8*hi
                float val = acc[g] + b3[s * COUT + o];
                x3s[((s * COUT + o) * 16 + tl) * 32 + v] = bf16rne(val);
            }
        }
    }
    __syncthreads();

    // ---- stage 2: z[t,u] = sum_v x3[o][t][v] * adj[o][u][v], summed over s ----
    for (int tile = wave; tile < 128; tile += 8) {
        int o  = tile >> 1;
        int ut = tile & 1;
        v8f acc = {};
        for (int s = 0; s < SS; ++s) {
            ABfrag a, b;
            const unsigned short* xrow = x3s + ((s * COUT + o) * 16 + lmod) * 32;
            int vb = lhi * 16;
            a.q[0] = *(const u32x4*)(xrow + vb);
            a.q[1] = *(const u32x4*)(xrow + 8 + vb);
            int u = ut * 16 + lmod;
            const unsigned short* brow =
                adjB + ((((size_t)s * NN + n) * COUT + o) * 1024) + u * 32 + lhi * 16;
            if (s + 1 < SS)
                __builtin_prefetch(brow + (size_t)NN * COUT * 1024, 0, 1);
            b.q[0] = *(const u32x4*)(brow);
            b.q[1] = *(const u32x4*)(brow + 8);
            acc = __builtin_amdgcn_wmma_f32_16x16x32_bf16(
                    false, a.v, false, b.v, (short)0, acc, false, false);
        }
        int u = ut * 16 + lmod;
        if (u < VV) {
            #pragma unroll
            for (int g = 0; g < 8; ++g) {
                int t = tb * 16 + g + lhi * 8;
                y[(((size_t)n * COUT + o) * TT + t) * VV + u] = acc[g];
            }
        }
    }
}

// ---------------- 4) BN statistics ----------------
__global__ void kzero(float* stats) { if (threadIdx.x < 128) stats[threadIdx.x] = 0.f; }

__global__ __launch_bounds__(256) void kstats(const float* __restrict__ y,
                                              float* __restrict__ stats) {
    __shared__ float r1[256], r2[256];
    int n = blockIdx.x >> 6, o = blockIdx.x & 63;
    const float* p = y + (((size_t)n * COUT + o) * TT) * VV;
    float s1 = 0.f, s2 = 0.f;
    for (int i = threadIdx.x; i < TT * VV; i += 256) {
        float v = p[i];
        s1 += v;
        s2 += v * v;
    }
    r1[threadIdx.x] = s1; r2[threadIdx.x] = s2;
    __syncthreads();
    for (int w = 128; w > 0; w >>= 1) {
        if (threadIdx.x < w) {
            r1[threadIdx.x] += r1[threadIdx.x + w];
            r2[threadIdx.x] += r2[threadIdx.x + w];
        }
        __syncthreads();
    }
    if (threadIdx.x == 0) {
        atomicAdd(&stats[o], r1[0]);
        atomicAdd(&stats[64 + o], r2[0]);
    }
}

// ---------------- 5) BN apply + residual + ReLU (in place on d_out) --------
__global__ __launch_bounds__(256) void kfinal(const float* __restrict__ x,
                                              const float* __restrict__ stats,
                                              const float* __restrict__ bnw,
                                              const float* __restrict__ bnb,
                                              float* __restrict__ y) {
    size_t id = (size_t)blockIdx.x * 256 + threadIdx.x;
    if (id >= (size_t)NN * COUT * TT * VV) return;
    int o = (int)((id / (TT * VV)) % COUT);
    const float cnt = (float)NN * TT * VV;   // 409600
    float m   = stats[o] / cnt;
    float var = stats[64 + o] / cnt - m * m;
    float val = (y[id] - m) * rsqrtf(var + 1e-5f) * bnw[o] + bnb[o] + x[id];
    y[id] = fmaxf(val, 0.f);
}

// ---------------- launcher ----------------
extern "C" void kernel_launch(void* const* d_in, const int* in_sizes, int n_in,
                              void* d_out, int out_size, void* d_ws, size_t ws_size,
                              hipStream_t stream) {
    (void)in_sizes; (void)n_in; (void)out_size; (void)ws_size;
    const float* x     = (const float*)d_in[0];
    const float* W1    = (const float*)d_in[1];
    const float* b1    = (const float*)d_in[2];
    const float* W2    = (const float*)d_in[3];
    const float* b2    = (const float*)d_in[4];
    const float* W3    = (const float*)d_in[5];
    const float* b3    = (const float*)d_in[6];
    const float* W4    = (const float*)d_in[7];
    const float* b4    = (const float*)d_in[8];
    const float* PA    = (const float*)d_in[9];
    const float* alpha = (const float*)d_in[10];
    const float* bnw   = (const float*)d_in[11];
    const float* bnb   = (const float*)d_in[12];
    float* out = (float*)d_out;

    // workspace carve (all 16B aligned)
    char* ws = (char*)d_ws;
    float* xbar = (float*)ws;                                  //  819,200 B
    unsigned short* adjB = (unsigned short*)(ws + 819200);     // 50,331,648 B
    float* stats = (float*)(ws + 819200 + 50331648);           //      512 B

    // allow 272KB dynamic LDS on the WMMA kernel (gfx1250 WGP has 320KB)
    static bool attr_done = false;
    if (!attr_done) {
        hipFuncSetAttribute((const void*)kmain,
                            hipFuncAttributeMaxDynamicSharedMemorySize, SM_TOT);
        attr_done = true;
    }

    kmean <<<(NN * CIN * VV + 255) / 256, 256, 0, stream>>>(x, xbar);
    kzero <<<1, 128, 0, stream>>>(stats);
    kadj  <<<SS * NN, 256, 0, stream>>>(xbar, W1, b1, W2, b2, W4, b4, PA, alpha, adjB);
    kmain <<<NN * (TT / 16), 256, SM_TOT, stream>>>(x, W3, b3, adjB, out);
    kstats<<<NN * COUT, 256, 0, stream>>>(out, stats);
    kfinal<<<(NN * COUT * TT * VV + 255) / 256, 256, 0, stream>>>(x, stats, bnw, bnb, out);
}
